// SILSTM_89902255439916
// MI455X (gfx1250) — matrix operand order repaired and checked by reading, not compiled, not run on hardware
//
#include <hip/hip_runtime.h>
#include <stdint.h>

// Problem dims (fixed by the reference)
#define T_STEPS 128
#define BATCH   64
#define DIM     1024
#define HID     1024
#define NLAYER  2
#define G4H     4096   // 4*HID
#define LN_EPS  1e-5f
#define NOISE_STD 0.1f
#define KT      64     // K chunk staged per LDS buffer

typedef __attribute__((ext_vector_type(16))) __bf16 v16bf;
typedef __attribute__((ext_vector_type(8)))  float  v8f;

// Async global->LDS path (gfx1250): guarded so the file compiles on any
// toolchain; fallback is a synchronous copy with identical barrier structure.
#if defined(__AMDGCN__) && \
    __has_builtin(__builtin_amdgcn_global_load_async_to_lds_b128) && \
    __has_builtin(__builtin_amdgcn_s_wait_asynccnt)
#define SILSTM_ASYNC 1
#else
#define SILSTM_ASYNC 0
#endif

#if SILSTM_ASYNC
// builtin expects int4-vector pointers: AS1 (printed '__device__') source,
// AS3 (printed '__shared__') LDS destination
typedef int silstm_v4i __attribute__((vector_size(16)));
#endif

// ---------------- cheap deterministic noise -------------------------------
__device__ __forceinline__ uint32_t hash32(uint32_t s) {
  s ^= s >> 17; s *= 0xed5ad4bbu;
  s ^= s >> 11; s *= 0xac4c1b51u;
  s ^= s >> 15; s *= 0x31848babu;
  s ^= s >> 14;
  return s;
}
__device__ __forceinline__ float noisef(uint32_t s) {
  // int32 -> uniform(-1,1) * sqrt(3)  => unit variance
  return (float)(int)hash32(s) * (1.7320508075688772f / 2147483648.0f);
}

// fp32 -> bf16 (round to nearest even)
__device__ __forceinline__ unsigned short f2bf(float f) {
  uint32_t u = __builtin_bit_cast(uint32_t, f);
  u += 0x7fffu + ((u >> 16) & 1u);
  return (unsigned short)(u >> 16);
}

union Frag16 { uint4 q[2]; v16bf v; };

// ---------------- tile staging (async DMA or sync fallback) ---------------
__device__ __forceinline__ void cp_b128(const unsigned short* __restrict__ g,
                                        unsigned short* l) {
#if SILSTM_ASYNC
  __builtin_amdgcn_global_load_async_to_lds_b128(
      (__attribute__((address_space(1))) silstm_v4i*)(g),
      (__attribute__((address_space(3))) silstm_v4i*)(l), 0, 0);
#else
  *(uint4*)l = *(const uint4*)g;
#endif
}

// Copy one 64xKT bf16 chunk of A and of W into LDS. 128 threads:
// 512 b128 per tile -> 4 per thread per tile -> 8 async ops per wave.
__device__ __forceinline__ void stage_chunk(const unsigned short* __restrict__ A,
                                            const unsigned short* __restrict__ W,
                                            unsigned short* lA, unsigned short* lW,
                                            int K, int n0, int k0, int tid) {
#pragma unroll
  for (int i = 0; i < 4; ++i) {
    const int q   = tid + i * 128;   // b128 index 0..511
    const int row = q >> 3;          // 0..63
    const int col = (q & 7) * 8;     // element column inside chunk
    cp_b128(A + (size_t)row * K + k0 + col, lA + q * 8);
    cp_b128(W + (size_t)(n0 + row) * K + k0 + col, lW + q * 8);
  }
}

// ---------------- WMMA GEMM:  C[64,N] = A[64,K] @ (W[N,K] + noise)^T ------
// 128 threads = 4 wave32.  Block tile M=64 x N=64; wave wid owns rows
// [wid*16, wid*16+16) across all 64 cols (4 16x16 WMMAs per 32-K slice).
// K staged in KT=64 chunks, double buffered, async copies overlap compute.
__launch_bounds__(128)
__global__ void silstm_gemm_wmma(const unsigned short* __restrict__ A,  // [64,K] bf16
                                 const unsigned short* __restrict__ W,  // [N,K] bf16
                                 float* __restrict__ Cout,              // [64,N] f32
                                 int K, int N, uint32_t seed) {
  __shared__ __align__(16) unsigned short lA[2][64 * KT];
  __shared__ __align__(16) unsigned short lW[2][64 * KT];

  const int tid  = threadIdx.x;
  const int lane = tid & 31;
  const int wid  = tid >> 5;
  const int m0   = wid * 16;
  const int n0   = blockIdx.x * 64;
  const int half = lane >> 4;
  const int l15  = lane & 15;
  const int nchunk = K / KT;

  v8f acc[4] = {};

  stage_chunk(A, W, lA[0], lW[0], K, n0, 0, tid);

  for (int c = 0; c < nchunk; ++c) {
    const int cur = c & 1;
    // issue next chunk into the other buffer (its last readers finished
    // before the trailing barrier of iteration c-1)
    if (c + 1 < nchunk)
      stage_chunk(A, W, lA[cur ^ 1], lW[cur ^ 1], K, n0, (c + 1) * KT, tid);
#if SILSTM_ASYNC
    // current chunk's 8 copies done; next chunk's 8 may remain in flight
    if (c + 1 < nchunk) __builtin_amdgcn_s_wait_asynccnt(8);
    else                __builtin_amdgcn_s_wait_asynccnt(0);
#endif
    __syncthreads();

    // ---- per-timestep weight noise, applied in LDS (2 bf16 per hash) ----
    {
      uint32_t* wb = (uint32_t*)lW[cur];
#pragma unroll
      for (int i = 0; i < 16; ++i) {
        const int p   = tid + i * 128;   // b32 index 0..2047
        const int nrw = p >> 5;          // weight row in chunk
        const int kp  = (p & 31) * 2;    // even k inside chunk
        const uint32_t idx = (uint32_t)((n0 + nrw) * K + c * KT + kp);
        const uint32_t h   = hash32(seed ^ (idx >> 1));
        const float sc = NOISE_STD * 1.7320508075688772f / 32768.0f;
        const float d0 = (float)(int16_t)(h & 0xffffu) * sc;
        const float d1 = (float)(int16_t)(h >> 16) * sc;
        const uint32_t w = wb[p];
        float w0 = __builtin_bit_cast(float, w << 16);
        float w1 = __builtin_bit_cast(float, w & 0xffff0000u);
        w0 += d0; w1 += d1;
        wb[p] = (uint32_t)f2bf(w0) | ((uint32_t)f2bf(w1) << 16);
      }
    }
    __syncthreads();

    // ---- fragments (ISA 7.12.2 16-bit layouts) + 8 WMMAs ----
    const unsigned short* As = lA[cur];
    const unsigned short* Ws = lW[cur];
#pragma unroll
    for (int s = 0; s < 2; ++s) {        // two 32-deep K slices per chunk
      Frag16 fa;
      const unsigned short* ab = As + (m0 + l15) * KT + s * 32;
      fa.q[0] = *(const uint4*)(ab + half * 8);        // K 0-7 / 8-15
      fa.q[1] = *(const uint4*)(ab + 16 + half * 8);   // K 16-23 / 24-31
#pragma unroll
      for (int g = 0; g < 4; ++g) {      // four 16-col groups
        Frag16 fb;
        const unsigned short* wp = Ws + (g * 16 + l15) * KT + s * 32;
        fb.q[0] = *(const uint4*)(wp + half * 8);
        fb.q[1] = *(const uint4*)(wp + 16 + half * 8);
        acc[g] = __builtin_amdgcn_wmma_f32_16x16x32_bf16(
            false, fa.v, false, fb.v, (short)0, acc[g], false, false);
      }
    }
    __syncthreads();   // protect buffers before they are restaged
  }

  // C/D layout: VGPR i -> (M = m0 + i + 8*half, N = l15)
  const int rowb = m0 + half * 8;
#pragma unroll
  for (int g = 0; g < 4; ++g) {
    const int col = n0 + g * 16 + l15;
#pragma unroll
    for (int i = 0; i < 8; ++i)
      Cout[(size_t)(rowb + i) * N + col] = acc[g][i];
  }
}

// ---------------- block reduction helper ----------------------------------
__device__ __forceinline__ float block_sum(float v, float* sbuf) {
#pragma unroll
  for (int off = 16; off; off >>= 1) v += __shfl_xor(v, off, 32);
  const int wid = threadIdx.x >> 5, lane = threadIdx.x & 31;
  if (lane == 0) sbuf[wid] = v;
  __syncthreads();
  float tot = 0.f;
#pragma unroll
  for (int i = 0; i < 8; ++i) tot += sbuf[i];
  __syncthreads();
  return tot;
}

__device__ __forceinline__ float sigmoidf_(float x) {
  return 1.0f / (1.0f + __expf(-x));
}

// ---------------- fused LN + gates + state update -------------------------
__launch_bounds__(256)
__global__ void silstm_pointwise(
    const float* __restrict__ a_ih,      // [B,4H]
    const float* __restrict__ a_hh,      // [B,4H]
    const float* __restrict__ bias_ih, const float* __restrict__ bias_hh,
    const float* __restrict__ ln_ih_w, const float* __restrict__ ln_ih_b,
    const float* __restrict__ ln_hh_w, const float* __restrict__ ln_hh_b,
    const float* __restrict__ ln_ho_w, const float* __restrict__ ln_ho_b,
    float* __restrict__ c_state,         // [B,H] f32 (in/out)
    unsigned short* __restrict__ h_state,// [B,H] bf16 (out; GEMM operand)
    float* __restrict__ y_out,           // [B,H] or null
    float* __restrict__ hT_out,          // [B,H] or null
    float* __restrict__ cT_out,          // [B,H] or null
    uint32_t seed_bih, uint32_t seed_bhh) {
  __shared__ float sred[8];
  const int b = blockIdx.x, tid = threadIdx.x;

  float vih[16], vhh[16];
  float s1 = 0.f, q1 = 0.f, s2 = 0.f, q2 = 0.f;
#pragma unroll
  for (int i = 0; i < 16; ++i) {
    const int e = tid + i * 256;
    float a  = a_ih[(size_t)b * G4H + e] + bias_ih[e]
             + NOISE_STD * noisef(seed_bih ^ (uint32_t)e);
    float hh = a_hh[(size_t)b * G4H + e] + bias_hh[e]
             + NOISE_STD * noisef(seed_bhh ^ (uint32_t)e);
    vih[i] = a;  vhh[i] = hh;
    s1 += a;  q1 += a * a;  s2 += hh; q2 += hh * hh;
  }
  s1 = block_sum(s1, sred); q1 = block_sum(q1, sred);
  s2 = block_sum(s2, sred); q2 = block_sum(q2, sred);
  const float m1 = s1 * (1.0f / G4H), m2 = s2 * (1.0f / G4H);
  const float r1 = rsqrtf(q1 * (1.0f / G4H) - m1 * m1 + LN_EPS);
  const float r2 = rsqrtf(q2 * (1.0f / G4H) - m2 * m2 + LN_EPS);

  float gates[16];
#pragma unroll
  for (int i = 0; i < 16; ++i) {
    const int e = tid + i * 256;
    gates[i] = ln_ih_w[e] * (vih[i] - m1) * r1 + ln_ih_b[e]
             + ln_hh_w[e] * (vhh[i] - m2) * r2 + ln_hh_b[e];
  }

  // e = tid + 256*i -> gate section i>>2, hidden index j = tid + 256*(i&3)
  float cn[4], og[4];
  float s3 = 0.f, q3 = 0.f;
#pragma unroll
  for (int qd = 0; qd < 4; ++qd) {
    const int j = tid + 256 * qd;
    const float ig = sigmoidf_(gates[qd]);
    const float fg = sigmoidf_(gates[4 + qd]);
    const float oo = sigmoidf_(gates[8 + qd]);
    const float gg = tanhf(gates[12 + qd]);
    const float cc = fg * c_state[(size_t)b * HID + j] + ig * gg;
    cn[qd] = cc; og[qd] = oo;
    s3 += cc; q3 += cc * cc;
  }
  s3 = block_sum(s3, sred); q3 = block_sum(q3, sred);
  const float m3 = s3 * (1.0f / HID);
  const float r3 = rsqrtf(q3 * (1.0f / HID) - m3 * m3 + LN_EPS);

#pragma unroll
  for (int qd = 0; qd < 4; ++qd) {
    const int j = tid + 256 * qd;
    const float hn = og[qd] * tanhf(ln_ho_w[j] * (cn[qd] - m3) * r3 + ln_ho_b[j]);
    c_state[(size_t)b * HID + j] = cn[qd];
    h_state[(size_t)b * HID + j] = f2bf(hn);
    if (y_out)  y_out [(size_t)b * HID + j] = hn;
    if (hT_out) hT_out[(size_t)b * HID + j] = hn;
    if (cT_out) cT_out[(size_t)b * HID + j] = cn[qd];
  }
}

// ---------------- small utility kernels ------------------------------------
__global__ void silstm_to_bf16(const float* __restrict__ src,
                               unsigned short* __restrict__ dst, int n) {
  const int i = blockIdx.x * blockDim.x + threadIdx.x;
  if (i < n) dst[i] = f2bf(src[i]);
}

__global__ void silstm_init_state(float* c, unsigned short* h, int n) {
  const int i = blockIdx.x * blockDim.x + threadIdx.x;
  if (i < n) { c[i] = 0.f; h[i] = 0; }
}

// ---------------- host driver ---------------------------------------------
extern "C" void kernel_launch(void* const* d_in, const int* in_sizes, int n_in,
                              void* d_out, int out_size, void* d_ws, size_t ws_size,
                              hipStream_t stream) {
  const float* x     = (const float*)d_in[0];   // [T,B,D]
  const float* Wih   = (const float*)d_in[1];   // [L,4H,D]
  const float* Whh   = (const float*)d_in[2];   // [L,4H,H]
  const float* bih   = (const float*)d_in[3];
  const float* bhh   = (const float*)d_in[4];
  const float* lnihw = (const float*)d_in[5];
  const float* lnihb = (const float*)d_in[6];
  const float* lnhhw = (const float*)d_in[7];
  const float* lnhhb = (const float*)d_in[8];
  const float* lnhow = (const float*)d_in[9];
  const float* lnhob = (const float*)d_in[10];
  float* out = (float*)d_out;

  const size_t LBH = (size_t)NLAYER * BATCH * HID;   // 131072
  const size_t NW  = (size_t)NLAYER * G4H * DIM;     // 8388608 (per weight tensor)
  const size_t NX  = (size_t)T_STEPS * BATCH * DIM;  // 8388608

  // workspace: f32 first, then bf16 tensors (~51 MB total)
  float* c_state = (float*)d_ws;                           // [L,B,H] f32
  float* a_ih    = c_state + LBH;                          // [B,4H]
  float* a_hh    = a_ih + (size_t)BATCH * G4H;
  unsigned short* x16   = (unsigned short*)(a_hh + (size_t)BATCH * G4H);
  unsigned short* Wih16 = x16 + NX;
  unsigned short* Whh16 = Wih16 + NW;
  unsigned short* h16   = Whh16 + NW;                      // [L,B,H] bf16

  // one-time (per launch) bf16 conversions: halves per-step HBM traffic
  silstm_to_bf16<<<(int)((NX + 255) / 256), 256, 0, stream>>>(x, x16, (int)NX);
  silstm_to_bf16<<<(int)((NW + 255) / 256), 256, 0, stream>>>(Wih, Wih16, (int)NW);
  silstm_to_bf16<<<(int)((NW + 255) / 256), 256, 0, stream>>>(Whh, Whh16, (int)NW);
  silstm_init_state<<<(int)((LBH + 255) / 256), 256, 0, stream>>>(c_state, h16, (int)LBH);

  float* yout = out;
  float* hT   = out + (size_t)T_STEPS * BATCH * HID;
  float* cT   = hT + LBH;

  for (int t = 0; t < T_STEPS; ++t) {
    for (int l = 0; l < NLAYER; ++l) {
      const uint32_t base = (uint32_t)(t * NLAYER + l) * 2654435761u;
      const unsigned short* A_in = (l == 0)
          ? (x16 + (size_t)t * BATCH * DIM)
          : (h16 + (size_t)(l - 1) * BATCH * HID);   // layer l-1 output at this t

      silstm_gemm_wmma<<<G4H / 64, 128, 0, stream>>>(
          A_in, Wih16 + (size_t)l * G4H * DIM, a_ih, DIM, G4H, base ^ 0x9e3779b9u);
      silstm_gemm_wmma<<<G4H / 64, 128, 0, stream>>>(
          h16 + (size_t)l * BATCH * HID, Whh16 + (size_t)l * G4H * HID,
          a_hh, HID, G4H, base ^ 0x7f4a7c15u);

      const bool last = (t == T_STEPS - 1);
      silstm_pointwise<<<BATCH, 256, 0, stream>>>(
          a_ih, a_hh,
          bih + (size_t)l * G4H, bhh + (size_t)l * G4H,
          lnihw + (size_t)l * G4H, lnihb + (size_t)l * G4H,
          lnhhw + (size_t)l * G4H, lnhhb + (size_t)l * G4H,
          lnhow + (size_t)l * HID, lnhob + (size_t)l * HID,
          c_state + (size_t)l * BATCH * HID, h16 + (size_t)l * BATCH * HID,
          (l == NLAYER - 1) ? (yout + (size_t)t * BATCH * HID) : nullptr,
          last ? (hT + (size_t)l * BATCH * HID) : nullptr,
          last ? (cT + (size_t)l * BATCH * HID) : nullptr,
          base ^ 0x85ebca6bu, base ^ 0xc2b2ae35u);
    }
  }
}
